// MHSelfAttention_42236708389096
// MI455X (gfx1250) — matrix operand order repaired and checked by reading, not compiled, not run on hardware
//
#include <hip/hip_runtime.h>
#include <hip/hip_bf16.h>
#include <stdint.h>

typedef __attribute__((ext_vector_type(16))) __bf16    v16bf;
typedef __attribute__((ext_vector_type(8)))  float     v8f;
typedef __attribute__((ext_vector_type(4)))  unsigned  u32x4;
typedef __attribute__((ext_vector_type(8)))  unsigned  u32x8;

#define N_EMBD    1024
#define N_HEAD    16
#define HEAD_DIM  64
#define BATCH     16
#define QLEN      32
#define CACHE_LEN 4064
#define FULL_LEN  4096
#define ROWS      (BATCH * QLEN)   // 512
#define LDS_ROW   68               // padded row stride (floats): 16 lanes -> 16 banks

// A-matrix (16x32 bf16) VGPR layout: element e=2j(+1) in lane-half hl holds
// K = 2*j + 16*hl (+1) (contiguous, matches B operands).
__device__ __forceinline__ int a_kbase(int j, int hl) {
  return (j << 1) + (hl << 4);
}

// Generic shared pointer -> wave-relative LDS byte offset (aperture: addr[31:0]).
__device__ __forceinline__ unsigned lds_off(const void* p) {
  return (unsigned)(uintptr_t)p;
}

__device__ __forceinline__ void wait_async0() {
  asm volatile("s_wait_asynccnt 0" ::: "memory");
}

// ---------------------------------------------------------------------------
// TDM: DMA one 32x64 f32 tile from global into LDS, padded +4 DWORDs per
// 64-DWORD row (pad_interval=5 -> 64 dw, pad_amount=3 -> 4 dw) => LDS_ROW=68.
// D# per CDNA5 ISA: group0 + group1, VADDR2/3 = NULL.
// ---------------------------------------------------------------------------
__device__ __forceinline__ void tdm_load_tile(unsigned lds_byte_addr,
                                              const float* gsrc) {
  const unsigned long long ga = (unsigned long long)(uintptr_t)gsrc;
  u32x4 g0;
  g0[0] = 1u;                                   // count=1, user mode, no gather
  g0[1] = lds_byte_addr;                        // lds_addr [63:32]
  g0[2] = (unsigned)ga;                         // global_addr[31:0]
  g0[3] = (unsigned)(ga >> 32) | (2u << 30);    // global_addr[56:32] | type=2
  u32x8 g1;
  g1[0] = (2u << 16) | (1u << 20) | (5u << 22) | (3u << 25); // 4B elems, padding
  g1[1] = (64u << 16);                          // tensor_dim0 = 64
  g1[2] = (32u << 16);                          // tensor_dim1 = 32
  g1[3] = (64u << 16);                          // tile_dim0 = 64
  g1[4] = 32u;                                  // tile_dim1 = 32
  g1[5] = 64u;                                  // tensor_dim0_stride = 64
  g1[6] = 0u;
  g1[7] = 0u;
  asm volatile("tensor_load_to_lds %0, %1" :: "s"(g0), "s"(g1) : "memory");
}

// Async copy of a 32x64 f32 tile LDS (row-padded) -> global, 64 thr, 8xB128/ln.
__device__ __forceinline__ void async_l2g_tile(float* __restrict__ gdst,
                                               unsigned lbase, int tid) {
#pragma unroll
  for (int i = 0; i < 8; ++i) {
    const int idx4 = i * 64 + tid;
    const int row  = idx4 >> 4;
    const int col4 = idx4 & 15;
    const unsigned laddr = lbase + row * (LDS_ROW * 4) + col4 * 16;
    const unsigned long long g =
        (unsigned long long)(uintptr_t)(gdst + ((long)row * HEAD_DIM + col4 * 4));
    asm volatile("global_store_async_from_lds_b128 %0, %1, off"
                 :: "v"(g), "v"(laddr) : "memory");
  }
}

// ---------------------------------------------------------------------------
// Row-major GEMM: C[M,N] = A[M,K]*B[K,N] + bias, bf16 WMMA, f32 accumulate.
// Each wave computes a 64x16 block (4 m-subtiles) so the strided B fragment
// is reused by 4 consecutive v_wmma. 8 waves / 256-thread block.
// ---------------------------------------------------------------------------
__global__ __launch_bounds__(256)
void gemm_bf16_wmma(const float* __restrict__ A, const float* __restrict__ Bm,
                    const float* __restrict__ bias, float* __restrict__ C,
                    int M, int N, int K) {
  const int lane = threadIdx.x & 31;
  const int wave = threadIdx.x >> 5;
  const int hl   = lane >> 4;
  const int lm   = lane & 15;
  const int tiles_n = N >> 4;
  const int tile = blockIdx.x * 8 + wave;
  if (tile >= (M >> 6) * tiles_n) return;
  const int mtb = tile / tiles_n;          // 64-row block index
  const int nt  = tile - mtb * tiles_n;
  const int n   = nt * 16 + lm;

  v8f acc[4] = {};
  for (int k0 = 0; k0 < K; k0 += 32) {
    v16bf bv;
#pragma unroll
    for (int e = 0; e < 16; ++e)           // B layout: K = e + 16*hl
      bv[e] = (__bf16)Bm[(long)(k0 + e + (hl << 4)) * N + n];
#pragma unroll
    for (int s = 0; s < 4; ++s) {
      v16bf av;
      const long m = (long)mtb * 64 + s * 16 + lm;
#pragma unroll
      for (int j = 0; j < 8; ++j) {
        const float* ap = A + m * K + k0 + a_kbase(j, hl);
        av[2 * j]     = (__bf16)ap[0];
        av[2 * j + 1] = (__bf16)ap[1];
      }
      acc[s] = __builtin_amdgcn_wmma_f32_16x16x32_bf16(false, av, false, bv,
                                                       (short)0, acc[s], false, false);
    }
  }
  const float bb = bias[n];
#pragma unroll
  for (int s = 0; s < 4; ++s)
#pragma unroll
    for (int r = 0; r < 8; ++r)            // C layout: row = base + r + 8*hl
      C[(long)(mtb * 64 + s * 16 + r + hl * 8) * N + n] = acc[s][r] + bb;
}

// ---------------------------------------------------------------------------
// Scatter the 32 newly-projected K/V rows into rows 4064..4095 of the outputs.
// ---------------------------------------------------------------------------
__global__ void scatter_new_kv(const float* __restrict__ qkv,
                               float* __restrict__ kout,
                               float* __restrict__ vout) {
  int i = blockIdx.x * blockDim.x + threadIdx.x;   // B*H*S*D = 524288
  if (i >= BATCH * N_HEAD * QLEN * HEAD_DIM) return;
  int d = i & 63;
  int s = (i >> 6) & 31;
  int h = (i >> 11) & 15;
  int b = i >> 15;
  long qrow = (long)(b * QLEN + s) * (3 * N_EMBD);
  float kv = qkv[qrow + N_EMBD + h * HEAD_DIM + d];
  float vv = qkv[qrow + 2 * N_EMBD + h * HEAD_DIM + d];
  long drow = ((long)(b * N_HEAD + h) * FULL_LEN + CACHE_LEN + s) * HEAD_DIM + d;
  kout[drow] = kv;
  vout[drow] = vv;
}

// ---------------------------------------------------------------------------
// Fused flash attention + cache-concat writeout. One workgroup (64 thr, 2
// waves) per (b,h). Per 32-key block: TDM-load K/V tiles into padded LDS,
// compute S^T = K*Q^T and O^T = V^T*P^T via bf16 WMMA, async-store the tile
// LDS -> concatenated k/v outputs. K/V touch HBM exactly once.
// ---------------------------------------------------------------------------
__global__ __launch_bounds__(64)
void flash_attn_fused(const float* __restrict__ qkv,
                      const float* __restrict__ kcache,
                      const float* __restrict__ vcache,
                      float* __restrict__ kout,
                      float* __restrict__ vout,
                      float* __restrict__ attn_out) {
  __shared__ float kbuf[32 * LDS_ROW];
  __shared__ float vbuf[32 * LDS_ROW];
  const int tid  = threadIdx.x;
  const int lane = tid & 31;
  const int qt   = tid >> 5;                 // wave id = query tile
  const int hl   = lane >> 4;
  const int lm   = lane & 15;
  const int h = blockIdx.x & (N_HEAD - 1);
  const int b = blockIdx.x >> 4;
  const int qloc = qt * 16 + lm;             // query 0..31 (this lane's column)
  const float scale = 0.125f;                // 1/sqrt(64), folded into Q

  const unsigned kl = lds_off(kbuf);
  const unsigned vl = lds_off(vbuf);
  const long bh = (long)(b * N_HEAD + h);
  const float* kc = kcache + bh * CACHE_LEN * HEAD_DIM;
  const float* vc = vcache + bh * CACHE_LEN * HEAD_DIM;
  float* ko = kout + bh * FULL_LEN * HEAD_DIM;
  float* vo = vout + bh * FULL_LEN * HEAD_DIM;

  // Q^T resident as two B-tiles (d-ranges 0..31, 32..63): B elem e -> d=e+16*hl
  const float* qp = qkv + (long)(b * QLEN + qloc) * (3 * N_EMBD) + h * HEAD_DIM;
  v16bf qb0, qb1;
#pragma unroll
  for (int e = 0; e < 16; ++e) {
    qb0[e] = (__bf16)(qp[e + (hl << 4)] * scale);
    qb1[e] = (__bf16)(qp[32 + e + (hl << 4)] * scale);
  }

  float mrun = -1e30f, lrun = 0.f;
  v8f o[4] = {};                             // O^T tiles: row d, col q=lm

  for (int it = 0; it < FULL_LEN / 32; ++it) {
    const int kb = it * 32;
    // Drain previous iteration's async stores so LDS can be overwritten.
    wait_async0();
    __syncthreads();
    const bool from_cache = (kb < CACHE_LEN);
    const float* ks = from_cache ? (kc + (long)kb * HEAD_DIM)
                                 : (ko + (long)kb * HEAD_DIM);  // new rows (scatter)
    const float* vs = from_cache ? (vc + (long)kb * HEAD_DIM)
                                 : (vo + (long)kb * HEAD_DIM);
    if (qt == 0) {                           // one wave drives the TDM
      tdm_load_tile(kl, ks);
      tdm_load_tile(vl, vs);
      __builtin_amdgcn_s_wait_tensorcnt(0);
    }
    __syncthreads();
    if (from_cache) {                        // fused cache-concat write-out
      async_l2g_tile(ko + (long)kb * HEAD_DIM, kl, tid);
      async_l2g_tile(vo + (long)kb * HEAD_DIM, vl, tid);
    }

    // ---- S^T (32 keys x 16 q) = K_blk * Q^T, two d-steps of 32 ----
    v8f s0 = {}, s1 = {};
#pragma unroll
    for (int ds = 0; ds < 2; ++ds) {
      v16bf a0, a1;                          // A = K rows (keys), K-dim = d
#pragma unroll
      for (int j = 0; j < 8; ++j) {
        const int dd = ds * 32 + a_kbase(j, hl);
        const float2 x0 = *(const float2*)&kbuf[lm * LDS_ROW + dd];
        const float2 x1 = *(const float2*)&kbuf[(16 + lm) * LDS_ROW + dd];
        a0[2 * j] = (__bf16)x0.x;  a0[2 * j + 1] = (__bf16)x0.y;
        a1[2 * j] = (__bf16)x1.x;  a1[2 * j + 1] = (__bf16)x1.y;
      }
      const v16bf qb = ds ? qb1 : qb0;
      s0 = __builtin_amdgcn_wmma_f32_16x16x32_bf16(false, a0, false, qb,
                                                   (short)0, s0, false, false);
      s1 = __builtin_amdgcn_wmma_f32_16x16x32_bf16(false, a1, false, qb,
                                                   (short)0, s1, false, false);
    }
    // ---- causal mask: only the final block (keys 4064..4095) ----
    if (it == FULL_LEN / 32 - 1) {
#pragma unroll
      for (int r = 0; r < 8; ++r) {
        if ((r + hl * 8) > qloc)      s0[r] = -1e30f;   // key_local = r+8*hl
        if ((16 + r + hl * 8) > qloc) s1[r] = -1e30f;
      }
    }
    // ---- online softmax (per-lane = per-query column) ----
    float bm = -1e30f;
#pragma unroll
    for (int r = 0; r < 8; ++r) bm = fmaxf(bm, fmaxf(s0[r], s1[r]));
    bm = fmaxf(bm, __shfl_xor(bm, 16));      // other 16 keys live in lane^16
    const float mnew  = fmaxf(mrun, bm);
    const float alpha = __expf(mrun - mnew);
    float rsum = 0.f;
    float p0[8], p1[8];
#pragma unroll
    for (int r = 0; r < 8; ++r) {
      p0[r] = __expf(s0[r] - mnew);
      p1[r] = __expf(s1[r] - mnew);
      rsum += p0[r] + p1[r];
    }
    rsum += __shfl_xor(rsum, 16);
    lrun = lrun * alpha + rsum;
    mrun = mnew;
#pragma unroll
    for (int dt = 0; dt < 4; ++dt)
#pragma unroll
      for (int r = 0; r < 8; ++r) o[dt][r] *= alpha;

    // ---- P^T into B layout (keys = e + 16*hl): one cross-half swap ----
    v16bf pb;
#pragma unroll
    for (int r = 0; r < 8; ++r) {
      const float send = hl ? p0[r] : p1[r];
      const float recv = __shfl_xor(send, 16);
      pb[r]     = (__bf16)(hl ? recv  : p0[r]);
      pb[r + 8] = (__bf16)(hl ? p1[r] : recv);
    }
    // ---- O^T += V^T * P^T : 4 d-tiles, K = 32 keys ----
#pragma unroll
    for (int dt = 0; dt < 4; ++dt) {
      v16bf va;                              // A = V^T: row d=dt*16+lm, k=key
#pragma unroll
      for (int j = 0; j < 8; ++j) {
        const int kk0 = a_kbase(j, hl);      // local key row in LDS tile
        va[2 * j]     = (__bf16)vbuf[kk0 * LDS_ROW + dt * 16 + lm];
        va[2 * j + 1] = (__bf16)vbuf[(kk0 + 1) * LDS_ROW + dt * 16 + lm];
      }
      o[dt] = __builtin_amdgcn_wmma_f32_16x16x32_bf16(false, va, false, pb,
                                                      (short)0, o[dt], false, false);
    }
  }
  wait_async0();                             // drain final write-out stores
  // ---- normalize and write attn output [512, 1024] row-major ----
  const float inv = 1.f / lrun;
  float* op = attn_out + (long)(b * QLEN + qloc) * N_EMBD + h * HEAD_DIM;
#pragma unroll
  for (int dt = 0; dt < 4; ++dt)
#pragma unroll
    for (int r = 0; r < 8; ++r)
      op[dt * 16 + r + hl * 8] = o[dt][r] * inv;
}

// ---------------------------------------------------------------------------
extern "C" void kernel_launch(void* const* d_in, const int* in_sizes, int n_in,
                              void* d_out, int out_size, void* d_ws, size_t ws_size,
                              hipStream_t stream) {
  (void)in_sizes; (void)n_in; (void)out_size; (void)ws_size;
  const float* hidden = (const float*)d_in[0];
  const float* kcache = (const float*)d_in[1];
  const float* vcache = (const float*)d_in[2];
  const float* w_qkv  = (const float*)d_in[3];
  const float* b_qkv  = (const float*)d_in[4];
  const float* w_out  = (const float*)d_in[5];
  const float* b_out  = (const float*)d_in[6];

  float* out  = (float*)d_out;                                    // [512,1024]
  float* kout = out + (long)ROWS * N_EMBD;                        // [B,H,4096,64]
  float* vout = kout + (long)BATCH * N_HEAD * FULL_LEN * HEAD_DIM;

  float* qkv_ws  = (float*)d_ws;                                  // [512,3072]
  float* attn_ws = qkv_ws + (long)ROWS * 3 * N_EMBD;              // [512,1024]

  // 1) fused QKV projection: [512,1024] x [1024,3072]
  {
    int tiles = (ROWS / 64) * (3 * N_EMBD / 16);  // 1536
    gemm_bf16_wmma<<<(tiles + 7) / 8, 256, 0, stream>>>(
        hidden, w_qkv, b_qkv, qkv_ws, ROWS, 3 * N_EMBD, N_EMBD);
  }
  // 2) append new K/V rows 4064..4095 to the concatenated outputs
  scatter_new_kv<<<(BATCH * N_HEAD * QLEN * HEAD_DIM + 255) / 256, 256, 0,
                   stream>>>(qkv_ws, kout, vout);
  // 3) fused flash attention + cache-concat stream (single HBM pass over K/V)
  flash_attn_fused<<<BATCH * N_HEAD, 64, 0, stream>>>(
      qkv_ws, kcache, vcache, kout, vout, attn_ws);
  // 4) output projection: [512,1024] x [1024,1024]
  {
    int tiles = (ROWS / 64) * (N_EMBD / 16);      // 512
    gemm_bf16_wmma<<<(tiles + 7) / 8, 256, 0, stream>>>(
        attn_ws, w_out, b_out, out, ROWS, N_EMBD, N_EMBD);
  }
}
